// NetworkLSTM_reimplement_28192165331637
// MI455X (gfx1250) — matrix operand order repaired and checked by reading, not compiled, not run on hardware
//
#include <hip/hip_runtime.h>
#include <hip/hip_bf16.h>
#include <cstdint>
#include <cstddef>

// ---------------------------------------------------------------------------
// Persistent-kernel LSTM for MI455X (gfx1250, wave32, WMMA).
//
// B=64, T=512, E=256, H=512, O=1. The T=512 recurrent scan is the critical
// path (each step: [64 x 768] . [768 x 2048], 201 MFLOP) and is latency-bound:
//   * ONE persistent kernel (128 single-wave workgroups) with a device-wide
//     atomic-counter barrier between steps -- no per-step launch latency.
//     All state (3 MB weights, 64 KB h, 16 MB x) is L2-resident (192 MB L2).
//   * ALL 96 weight B-fragments (24 k-steps x 4 gates) loaded ONCE into the
//     extended 1024-VGPR file (768 VGPRs) -- zero per-step weight traffic.
//   * x-region A-fragments for step t+1 preloaded into registers BEFORE the
//     barrier (independent of h) -- the 32 x-part WMMAs run wait-free.
//   * h-region A-fragments: HPIPE-deep software pipeline. The pre-RA
//     scheduler was sinking the loads to depth-1 under register pressure
//     (round-4 asm: one recycled buffer + s_wait_loadcnt 0x0 per k-step), so
//     __builtin_amdgcn_sched_barrier(0) fences pin each load ABOVE the
//     preceding WMMA group, keeping HPIPE loads in flight.
//   * h ping/pong selected by OFFSET from one base pointer -> global_load.
//   * c-state register-resident for all 512 steps; only h round-trips L2.
// ---------------------------------------------------------------------------

typedef __bf16 bf16_t;
typedef __attribute__((ext_vector_type(16))) __bf16 v16bf;
typedef __attribute__((ext_vector_type(8)))  __bf16 v8bf;
typedef __attribute__((ext_vector_type(8)))  float  v8f;

#define B_     64
#define T_     512
#define E_     256
#define H_     512
#define KTOT   (E_ + H_)        // 768
#define NKS    (KTOT / 32)      // 24 k-steps (8 x-region + 16 h-region)
#define NXS    (E_ / 32)        // 8 x-region k-steps
#define NHS    (H_ / 32)        // 16 h-region k-steps
#define NROW   (4 * H_)         // 2048 gate rows
#define NBLK   128              // 4 batch-blocks x 32 h-blocks, 1 wave each
#define HPIPE  5                // h-fragment software-pipeline depth
#define BH     (B_ * H_)        // one h buffer

#if __has_builtin(__builtin_amdgcn_sched_barrier)
#define SCHED_FENCE() __builtin_amdgcn_sched_barrier(0)
#else
#define SCHED_FENCE()
#endif

__device__ __forceinline__ float fsig(float x) { return 1.0f / (1.0f + __expf(-x)); }

// assemble a v16bf fragment from 16 consecutive bf16
__device__ __forceinline__ v16bf ld_frag16(const bf16_t* p)
{
    v8bf lo = *(const v8bf*)(p);
    v8bf hi = *(const v8bf*)(p + 8);
    return __builtin_shufflevector(lo, hi, 0,1,2,3,4,5,6,7,8,9,10,11,12,13,14,15);
}

// bf16 A-fragment (16x32, row-major source): lane l, row = l&15, half = l>>4:
//   e[0..7] = K[8*half .. +7], e[8..15] = K[16 + 8*half .. +7]
__device__ __forceinline__ v16bf ld_afrag(const bf16_t* row, int half)
{
    v8bf lo = *(const v8bf*)(row + 8 * half);
    v8bf hi = *(const v8bf*)(row + 16 + 8 * half);
    return __builtin_shufflevector(lo, hi, 0,1,2,3,4,5,6,7,8,9,10,11,12,13,14,15);
}

// ---- build fragment-ordered bf16 weights:
//   Wf[(((hblk*NKS + ks)*4 + g)*32 + lane)*16 + i]
//     = W[row = g*H + hblk*16 + (lane&15)][k = ks*32 + 8*(lane>>4) + (i&7) + 16*(i>>3)]
// where W = [Wi | Wh], K = E+H. One contiguous 1KB block per (wave,ks,g).
__global__ __launch_bounds__(256)
void k_build_wfrag(const float* __restrict__ Wi, const float* __restrict__ Wh,
                   bf16_t* __restrict__ Wf)
{
    size_t idx = (size_t)blockIdx.x * blockDim.x + threadIdx.x;   // NROW*KTOT
    if (idx >= (size_t)NROW * KTOT) return;
    int i    = (int)(idx & 15);
    int lane = (int)((idx >> 4) & 31);
    int g    = (int)((idx >> 9) & 3);
    int hks  = (int)(idx >> 11);          // hblk*NKS + ks
    int ks   = hks % NKS;
    int hblk = hks / NKS;
    int row  = g * H_ + hblk * 16 + (lane & 15);
    int k    = ks * 32 + 8 * (lane >> 4) + (i & 7) + 16 * (i >> 3);
    float w  = (k < E_) ? Wi[(size_t)row * E_ + k]
                        : Wh[(size_t)row * H_ + (k - E_)];
    Wf[idx] = (bf16_t)w;
}

// ---- combined bias bi+bh (fp32, 2048)
__global__ __launch_bounds__(256)
void k_bias(const float* __restrict__ bi, const float* __restrict__ bh,
            float* __restrict__ bias)
{
    int i = blockIdx.x * blockDim.x + threadIdx.x;
    if (i < NROW) bias[i] = bi[i] + bh[i];
}

// ---- embedding gather + fp32->bf16: xbf[b][t][e]
__global__ __launch_bounds__(256)
void k_embed(const int* __restrict__ words, const float* __restrict__ emb,
             bf16_t* __restrict__ xbf)
{
    size_t idx = (size_t)blockIdx.x * blockDim.x + threadIdx.x;   // B*T*E
    if (idx >= (size_t)B_ * T_ * E_) return;
    int e  = (int)(idx % E_);
    int bt = (int)(idx / E_);            // b*T + t
    int w  = words[bt];
    xbf[idx] = (bf16_t)emb[(size_t)w * E_ + e];
}

// ---- zero h ping buffer (bf16) and the barrier counter
__global__ __launch_bounds__(256)
void k_zero_state(bf16_t* __restrict__ h0, unsigned* __restrict__ ctr)
{
    int i = blockIdx.x * blockDim.x + threadIdx.x;
    if (i < BH) h0[i] = (bf16_t)0.0f;
    if (i == 0) *ctr = 0u;
}

// ---------------------------------------------------------------------------
// Persistent scan: grid = 128 blocks x 32 threads (one full wave each, EXEC
// all-ones as WMMA requires). Loops over all T=512 timesteps internally.
// ---------------------------------------------------------------------------
__global__ __launch_bounds__(32)
void k_scan(const bf16_t* __restrict__ xbf,     // [B][T][E]
            const bf16_t* __restrict__ Wf,      // fragment-ordered weights
            const float*  __restrict__ bias,    // [2048]
            bf16_t*       __restrict__ hball,   // [2][B][H] ping+pong (contig)
            float*        __restrict__ hout,    // [B][H] fp32 (final h)
            unsigned*     __restrict__ ctr)     // barrier counter
{
    const int lane = threadIdx.x;
    const int half = lane >> 4;
    const int l16  = lane & 15;
    const int bblk = blockIdx.x & 3;     // batch block (4 x 16 = 64)
    const int hblk = blockIdx.x >> 2;    // h block    (32 x 16 = 512)

    const int arow = bblk * 16 + l16;    // batch row this lane loads for A
    const int ncol = hblk * 16 + l16;    // h column this lane owns in C

    // bias: loaded once, register-resident
    float bv[4];
#pragma unroll
    for (int g = 0; g < 4; ++g) bv[g] = bias[g * H_ + ncol];

    // c-state: 8 cells per lane, register-resident for all 512 steps
    float creg[8];
#pragma unroll
    for (int v = 0; v < 8; ++v) creg[v] = 0.0f;

    // ---- ALL weight fragments into registers (768 VGPRs), once
    const bf16_t* wbase = Wf + ((size_t)hblk * NKS * 4) * 512 + (size_t)lane * 16;
    v16bf wreg[NKS][4];
#pragma unroll
    for (int ks = 0; ks < NKS; ++ks)
#pragma unroll
        for (int g = 0; g < 4; ++g)
            wreg[ks][g] = ld_frag16(wbase + ((size_t)ks * 4 + g) * 512);

    // ---- preload x-region A-fragments for t = 0
    v16bf xf[NXS];
    {
        const bf16_t* xr0 = xbf + ((size_t)arow * T_) * E_;
#pragma unroll
        for (int i = 0; i < NXS; ++i) xf[i] = ld_afrag(xr0 + 32 * i, half);
    }

    for (int t = 0; t < T_; ++t) {
        // ping/pong by OFFSET from one global base -> global_load, not flat
        const bf16_t* hrow  = hball + (size_t)(t & 1) * BH + (size_t)arow * H_;
        bf16_t*       hnext = hball + (size_t)((t + 1) & 1) * BH;

        v8f acc[4];
#pragma unroll
        for (int g = 0; g < 4; ++g)
#pragma unroll
            for (int v = 0; v < 8; ++v) acc[g][v] = bv[g];

        // h-fragments: fresh SSA value per fragment; issue the first HPIPE
        // loads now (post-acquire) -- the x-part WMMAs below are their cover.
        // The sched fence keeps them from sinking below the x-part WMMAs.
        v16bf hfa[NHS];
#pragma unroll
        for (int p = 0; p < HPIPE; ++p) hfa[p] = ld_afrag(hrow + 32 * p, half);
        SCHED_FENCE();

        // ---- x-region k-steps: A-frags already in registers (preloaded
        //      during the previous barrier wait)
#pragma unroll
        for (int ks = 0; ks < NXS; ++ks)
#pragma unroll
            for (int g = 0; g < 4; ++g)
                acc[g] = __builtin_amdgcn_wmma_f32_16x16x32_bf16(
                             false, xf[ks], false, wreg[ks][g],
                             (short)0, acc[g], false, false);

        // ---- h-region k-steps: HPIPE-deep pipeline. The fence after each
        //      lookahead load pins it ABOVE this iteration's WMMA group, so
        //      HPIPE fragment loads stay in flight (partial waitcnts, not 0).
#pragma unroll
        for (int j = 0; j < NHS; ++j) {
            if (j + HPIPE < NHS) {
                hfa[j + HPIPE] = ld_afrag(hrow + 32 * (j + HPIPE), half);
                SCHED_FENCE();
            }
#pragma unroll
            for (int g = 0; g < 4; ++g)
                acc[g] = __builtin_amdgcn_wmma_f32_16x16x32_bf16(
                             false, hfa[j], false, wreg[NXS + j][g],
                             (short)0, acc[g], false, false);
        }

        // ---- gate nonlinearities + state update (C layout: M = v + 8*half)
#pragma unroll
        for (int v = 0; v < 8; ++v) {
            int brow   = bblk * 16 + v + 8 * half;
            size_t idx = (size_t)brow * H_ + ncol;
            float r  = fsig(acc[0][v]);
            float f  = fsig(acc[1][v]);
            float gg = tanhf(acc[2][v]);
            float o  = fsig(acc[3][v]);
            float cn = f * creg[v] + r * gg;
            creg[v] = cn;
            float h = o * tanhf(cn);
            hnext[idx] = (bf16_t)h;
            if (t == T_ - 1) hout[idx] = h;
        }

        // ---- preload next step's x-region A-fragments; their L2 latency is
        //      fully hidden behind the barrier wait below
        if (t + 1 < T_) {
            const bf16_t* xn = xbf + ((size_t)arow * T_ + (t + 1)) * E_;
#pragma unroll
            for (int i = 0; i < NXS; ++i) xf[i] = ld_afrag(xn + 32 * i, half);
        }

        // ---- device-wide barrier: monotonic counter, release/acquire
        __threadfence();                               // publish h stores (agent)
        if (lane == 0) {
            __hip_atomic_fetch_add(ctr, 1u, __ATOMIC_ACQ_REL, __HIP_MEMORY_SCOPE_AGENT);
            const unsigned tgt = (unsigned)NBLK * (unsigned)(t + 1);
            while (__hip_atomic_load(ctr, __ATOMIC_ACQUIRE, __HIP_MEMORY_SCOPE_AGENT) < tgt)
                __builtin_amdgcn_s_sleep(4);
        }
        __threadfence();                               // acquire for all lanes
    }
}

// ---- out[b] = sigmoid(h[b] . Wfc + bfc), O = 1
__global__ __launch_bounds__(256)
void k_fc(const float* __restrict__ h, const float* __restrict__ Wfc,
          const float* __restrict__ bfc, float* __restrict__ out)
{
    __shared__ float red[256];
    int b = blockIdx.x;
    float s = 0.0f;
    for (int n = threadIdx.x; n < H_; n += 256)
        s += h[(size_t)b * H_ + n] * Wfc[n];
    red[threadIdx.x] = s;
    __syncthreads();
    for (int st = 128; st > 0; st >>= 1) {
        if (threadIdx.x < st) red[threadIdx.x] += red[threadIdx.x + st];
        __syncthreads();
    }
    if (threadIdx.x == 0)
        out[b] = 1.0f / (1.0f + __expf(-(red[0] + bfc[0])));
}

// ---------------------------------------------------------------------------
extern "C" void kernel_launch(void* const* d_in, const int* in_sizes, int n_in,
                              void* d_out, int out_size, void* d_ws, size_t ws_size,
                              hipStream_t stream)
{
    const int*   words = (const int*)  d_in[0];   // [B,T]
    const float* emb   = (const float*)d_in[1];   // [V,E]
    const float* Wi    = (const float*)d_in[2];   // [4,H,E]
    const float* bi    = (const float*)d_in[3];   // [4,H]
    const float* Wh    = (const float*)d_in[4];   // [4,H,H]
    const float* bh    = (const float*)d_in[5];   // [4,H]
    const float* Wfc   = (const float*)d_in[6];   // [1,H]
    const float* bfc   = (const float*)d_in[7];   // [1]

    float* out  = (float*)d_out;       // [64] out, then [64*512] h (fp32)
    float* hout = out + B_;

    // workspace layout (256B-aligned), ~19.1 MB total
    char* ws = (char*)d_ws;
    auto align256 = [](size_t x) { return (x + 255) & ~(size_t)255; };
    size_t off = 0;
    bf16_t*   Wf    = (bf16_t*)  (ws + off); off = align256(off + (size_t)NROW * KTOT * sizeof(bf16_t));
    float*    bias  = (float*)   (ws + off); off = align256(off + (size_t)NROW * sizeof(float));
    bf16_t*   xbf   = (bf16_t*)  (ws + off); off = align256(off + (size_t)B_ * T_ * E_ * sizeof(bf16_t));
    bf16_t*   hball = (bf16_t*)  (ws + off); off = align256(off + (size_t)2 * BH * sizeof(bf16_t));
    unsigned* ctr   = (unsigned*)(ws + off); off = align256(off + 256);
    (void)ws_size; (void)in_sizes; (void)n_in; (void)out_size;

    // prep: fragment-ordered weights, bias, embedding gather, state zero
    {
        size_t n = (size_t)NROW * KTOT;
        k_build_wfrag<<<(unsigned)((n + 255) / 256), 256, 0, stream>>>(Wi, Wh, Wf);
    }
    k_bias<<<(NROW + 255) / 256, 256, 0, stream>>>(bi, bh, bias);
    {
        size_t n = (size_t)B_ * T_ * E_;
        k_embed<<<(unsigned)((n + 255) / 256), 256, 0, stream>>>(words, emb, xbf);
    }
    k_zero_state<<<(BH + 255) / 256, 256, 0, stream>>>(hball, ctr);

    // one persistent kernel for the whole sequential scan
    k_scan<<<NBLK, 32, 0, stream>>>(xbf, Wf, bias, hball, hout, ctr);

    // final projection
    k_fc<<<B_, 256, 0, stream>>>(hout, Wfc, bfc, out);
}
